// SmallAttention_34411277976107
// MI455X (gfx1250) — matrix operand order repaired and checked by reading, not compiled, not run on hardware
//
#include <hip/hip_runtime.h>

// MI455X / gfx1250, wave32. Flash attention in bf16 WMMA (f32 accumulate).
//   B=4, S=4096, D_IN=512, D_HEAD=64, D_OUT=512, NQKV=192
// ws layout: q[2MB] | k[2MB] | vT[2MB] | o[2MB] | WqkvT bf16 [192][512] (192KB)
//            | WoutT bf16 [512][64] (64KB)

typedef __bf16 bf16;
typedef __attribute__((ext_vector_type(16))) __bf16 v16bf;
typedef __attribute__((ext_vector_type(8)))  __bf16 v8bf;
typedef __attribute__((ext_vector_type(8)))  float  v8f;

#define WMMA_BF16(a, b, c) \
    __builtin_amdgcn_wmma_f32_16x16x32_bf16(false, (a), false, (b), (short)0, (c), false, false)

// ---------------------------------------------------------------------------
// Kernel 0: convert + transpose weights to bf16 so WMMA B-operands are
// contiguous 32B loads. WqkvT[n][k] (192x512), WoutT[n][k] (512x64).
// ---------------------------------------------------------------------------
__global__ void __launch_bounds__(256) prep_weights_kernel(
    const float* __restrict__ Wqkv, const float* __restrict__ Wout,
    bf16* __restrict__ wqkvT, bf16* __restrict__ woutT)
{
    const int N1 = 512 * 192;                       // Wqkv elements
    const int N2 = 64 * 512;                        // Wout elements
    for (int idx = blockIdx.x * 256 + threadIdx.x; idx < N1 + N2;
         idx += gridDim.x * 256) {
        if (idx < N1) {
            const int kk = idx / 192, nn = idx % 192;
            wqkvT[(size_t)nn * 512 + kk] = (bf16)Wqkv[idx];
        } else {
            const int i2 = idx - N1;
            const int kk = i2 / 512, nn = i2 % 512;
            woutT[(size_t)nn * 64 + kk] = (bf16)Wout[i2];
        }
    }
}

// ---------------------------------------------------------------------------
// Kernel 1: qkv = x @ W_qkv ; write q*scale, k (row-major) and v (transposed)
// One wave = one 16x16 output tile, K=512 in 16 WMMA steps.
// A: 4x global_load_b128 fp32 + cvt_pk_bf16. B: 2x global_load_b128 (L2-hot).
// ---------------------------------------------------------------------------
__global__ void __launch_bounds__(256) qkv_proj_kernel(
    const float* __restrict__ x, const bf16* __restrict__ wqkvT,
    bf16* __restrict__ qb, bf16* __restrict__ kb, bf16* __restrict__ vtb)
{
    const int wave = threadIdx.x >> 5;
    const int lane = threadIdx.x & 31;
    const int g    = lane >> 4;
    const int hn   = lane & 15;

    const int wg    = blockIdx.x * 8 + wave;        // 0..12287
    const int ntile = wg % 12;
    const int mtile = wg / 12;
    const int row0  = mtile * 16;
    const int ncol  = ntile * 16 + hn;

    v8f acc = {};
    const float* xr = x + (size_t)(row0 + hn) * 512;
    const bf16*  wr = wqkvT + (size_t)ncol * 512;
#pragma unroll
    for (int kt = 0; kt < 16; ++kt) {
        const int k0 = kt * 32;
        const float4 s0a = *(const float4*)(xr + k0 + g * 16);
        const float4 s0b = *(const float4*)(xr + k0 + g * 16 + 4);
        const float4 s1a = *(const float4*)(xr + k0 + g * 16 + 8);
        const float4 s1b = *(const float4*)(xr + k0 + g * 16 + 12);
        v16bf a;
        a[0]  = (bf16)s0a.x; a[1]  = (bf16)s0a.y; a[2]  = (bf16)s0a.z; a[3]  = (bf16)s0a.w;
        a[4]  = (bf16)s0b.x; a[5]  = (bf16)s0b.y; a[6]  = (bf16)s0b.z; a[7]  = (bf16)s0b.w;
        a[8]  = (bf16)s1a.x; a[9]  = (bf16)s1a.y; a[10] = (bf16)s1a.z; a[11] = (bf16)s1a.w;
        a[12] = (bf16)s1b.x; a[13] = (bf16)s1b.y; a[14] = (bf16)s1b.z; a[15] = (bf16)s1b.w;
        const v16bf b = *(const v16bf*)(wr + k0 + g * 16);   // contiguous B-layout
        acc = WMMA_BF16(a, b, acc);
    }

#pragma unroll
    for (int v = 0; v < 8; ++v) {
        const int r = row0 + g * 8 + v;             // global row in [0,16384)
        const float val = acc[v];
        if (ncol < 64) {
            qb[(size_t)r * 64 + ncol] = (bf16)(val * 0.125f);   // fold 1/sqrt(64)
        } else if (ncol < 128) {
            kb[(size_t)r * 64 + (ncol - 64)] = (bf16)val;
        } else {
            const int d = ncol - 128;
            const int bidx = r >> 12, s = r & 4095;
            vtb[((size_t)bidx * 64 + d) * 4096 + s] = (bf16)val;  // v transposed
        }
    }
}

// ---------------------------------------------------------------------------
// Kernel 2: causal flash attention. One wave = one 16-row query tile.
// Key loop in blocks of 32; per block: 4 WMMAs for scores, online softmax,
// P transposed through per-wave LDS, 4 WMMAs for P@V (d_head = 4 N-tiles).
// ---------------------------------------------------------------------------
__global__ void __launch_bounds__(256) attn_kernel(
    const bf16* __restrict__ qb, const bf16* __restrict__ kb,
    const bf16* __restrict__ vtb, bf16* __restrict__ ob)
{
    __shared__ __align__(32) bf16 plds[8][16 * 32];
    const int wave = threadIdx.x >> 5;
    const int lane = threadIdx.x & 31;
    const int g    = lane >> 4;
    const int hn   = lane & 15;

    const int qt   = blockIdx.x * 8 + wave;         // 0..1023
    const int bidx = qt >> 8;
    const int i0   = (qt & 255) << 4;
    const size_t qrow = (size_t)bidx * 4096 + i0;

    // Q tile in A layout, both 32-wide d chunks, kept in registers
    v16bf qa[2];
    {
        const bf16* qp = qb + (qrow + hn) * 64;
#pragma unroll
        for (int dc = 0; dc < 2; ++dc) {
            const v8bf lo = *(const v8bf*)(qp + dc * 32 + g * 16);
            const v8bf hi = *(const v8bf*)(qp + dc * 32 + g * 16 + 8);
#pragma unroll
            for (int e = 0; e < 8; ++e) { qa[dc][e] = lo[e]; qa[dc][8 + e] = hi[e]; }
        }
    }

    v8f acc[4] = {};
    float mrun[8], lrun[8];
#pragma unroll
    for (int v = 0; v < 8; ++v) { mrun[v] = -1e30f; lrun[v] = 0.f; }

    const int nj = (i0 + 16 + 31) >> 5;             // causal: only blocks with j0 <= i0+15
    const bf16* kbase = kb  + (size_t)bidx * 4096 * 64;
    const bf16* vbase = vtb + (size_t)bidx * 64 * 4096;
    bf16* myp = &plds[wave][0];

    for (int jb = 0; jb < nj; ++jb) {
        const int j0 = jb * 32;

        // S = (q*scale) @ k^T : two 16-key subtiles x two 32-wide d chunks
        v8f s0 = {}, s1 = {};
        {
            const bf16* kp0 = kbase + (size_t)(j0 + hn) * 64;
            const bf16* kp1 = kbase + (size_t)(j0 + 16 + hn) * 64;
#pragma unroll
            for (int dc = 0; dc < 2; ++dc) {
                const v16bf b0 = *(const v16bf*)(kp0 + dc * 32 + g * 16);
                const v16bf b1 = *(const v16bf*)(kp1 + dc * 32 + g * 16);
                s0 = WMMA_BF16(qa[dc], b0, s0);
                s1 = WMMA_BF16(qa[dc], b1, s1);
            }
        }
        if (jb + 1 < nj)                             // global_prefetch_b8 of next K block
            __builtin_prefetch(kbase + (size_t)(j0 + 32 + lane) * 64, 0, 1);

        const bool need_mask = (j0 + 31 > i0);

        // online softmax, per owned row (rows 8g+v live across lanes of this half-group)
#pragma unroll
        for (int v = 0; v < 8; ++v) {
            const int irow = i0 + g * 8 + v;
            float a0 = s0[v], a1 = s1[v];
            if (need_mask) {
                if (j0 + hn      > irow) a0 = -1e30f;
                if (j0 + 16 + hn > irow) a1 = -1e30f;
            }
            float bm = fmaxf(a0, a1);
#pragma unroll
            for (int m = 8; m >= 1; m >>= 1) bm = fmaxf(bm, __shfl_xor(bm, m, 32));
            const float mnew = fmaxf(mrun[v], bm);
            const float corr = __expf(mrun[v] - mnew);
            mrun[v] = mnew;
            const float p0 = __expf(a0 - mnew);
            const float p1 = __expf(a1 - mnew);
            float rs = p0 + p1;
#pragma unroll
            for (int m = 8; m >= 1; m >>= 1) rs += __shfl_xor(rs, m, 32);
            lrun[v] = lrun[v] * corr + rs;
#pragma unroll
            for (int t = 0; t < 4; ++t) acc[t][v] *= corr;
            myp[(g * 8 + v) * 32 + hn]      = (bf16)p0;   // C-layout -> LDS [row][col]
            myp[(g * 8 + v) * 32 + 16 + hn] = (bf16)p1;
        }

        // read P back in A layout (ds_load_b128 x2), then P @ V over 4 d-tiles
        v16bf pa;
        {
            const v8bf lo = *(const v8bf*)(myp + hn * 32 + g * 16);
            const v8bf hi = *(const v8bf*)(myp + hn * 32 + g * 16 + 8);
#pragma unroll
            for (int e = 0; e < 8; ++e) { pa[e] = lo[e]; pa[8 + e] = hi[e]; }
        }
#pragma unroll
        for (int t = 0; t < 4; ++t) {
            const v16bf vbm = *(const v16bf*)(vbase + (size_t)(t * 16 + hn) * 4096 + j0 + g * 16);
            acc[t] = WMMA_BF16(pa, vbm, acc[t]);
        }
    }

    // normalize and store O (bf16, row-major [S][64])
#pragma unroll
    for (int v = 0; v < 8; ++v) {
        const float inv = 1.f / lrun[v];
        bf16* op = ob + (qrow + g * 8 + v) * 64;
#pragma unroll
        for (int t = 0; t < 4; ++t) op[t * 16 + hn] = (bf16)(acc[t][v] * inv);
    }
}

// ---------------------------------------------------------------------------
// Kernel 3: out = o @ W_out + b_out. One wave = one 16x16 tile, K=64 (2 steps).
// ---------------------------------------------------------------------------
__global__ void __launch_bounds__(256) out_proj_kernel(
    const bf16* __restrict__ ob, const bf16* __restrict__ woutT,
    const float* __restrict__ bout, float* __restrict__ out)
{
    const int wave = threadIdx.x >> 5;
    const int lane = threadIdx.x & 31;
    const int g    = lane >> 4;
    const int hn   = lane & 15;

    const int wg    = blockIdx.x * 8 + wave;        // 0..32767
    const int ntile = wg & 31;
    const int mtile = wg >> 5;
    const int row0  = mtile * 16;
    const int ncol  = ntile * 16 + hn;

    v8f acc = {};
    const bf16* orow = ob + (size_t)(row0 + hn) * 64;
    const bf16* wr   = woutT + (size_t)ncol * 64;
#pragma unroll
    for (int kt = 0; kt < 2; ++kt) {
        const int k0 = kt * 32;
        const v8bf lo = *(const v8bf*)(orow + k0 + g * 16);
        const v8bf hi = *(const v8bf*)(orow + k0 + g * 16 + 8);
        v16bf a;
#pragma unroll
        for (int e = 0; e < 8; ++e) { a[e] = lo[e]; a[8 + e] = hi[e]; }
        const v16bf b = *(const v16bf*)(wr + k0 + g * 16);   // contiguous B-layout
        acc = WMMA_BF16(a, b, acc);
    }

    const float bias = bout[ncol];
#pragma unroll
    for (int v = 0; v < 8; ++v)
        out[(size_t)(row0 + g * 8 + v) * 512 + ncol] = acc[v] + bias;
}

// ---------------------------------------------------------------------------
extern "C" void kernel_launch(void* const* d_in, const int* in_sizes, int n_in,
                              void* d_out, int out_size, void* d_ws, size_t ws_size,
                              hipStream_t stream) {
    const float* x    = (const float*)d_in[0];   // [4,4096,512]
    const float* Wqkv = (const float*)d_in[1];   // [512,192]
    const float* Wout = (const float*)d_in[2];   // [64,512]
    const float* bout = (const float*)d_in[3];   // [512]
    float* out = (float*)d_out;                  // [4,4096,512]

    char* ws = (char*)d_ws;
    const size_t tsz = (size_t)4 * 4096 * 64 * sizeof(bf16);   // 2 MB per tensor
    bf16* qb     = (bf16*)(ws);
    bf16* kb     = (bf16*)(ws + tsz);
    bf16* vtb    = (bf16*)(ws + 2 * tsz);
    bf16* obf    = (bf16*)(ws + 3 * tsz);
    bf16* wqkvT  = (bf16*)(ws + 4 * tsz);                       // 192*512*2 = 192KB
    bf16* woutT  = (bf16*)(ws + 4 * tsz + (size_t)192 * 512 * 2);  // 512*64*2 = 64KB

    prep_weights_kernel<<<64,       256, 0, stream>>>(Wqkv, Wout, wqkvT, woutT);
    qkv_proj_kernel    <<<128 * 12, 256, 0, stream>>>(x, wqkvT, qb, kb, vtb);
    attn_kernel        <<<128,      256, 0, stream>>>(qb, kb, vtb, obf);
    out_proj_kernel    <<<128 * 32, 256, 0, stream>>>(obf, woutT, bout, out);
}
